// my_multi_head_atten_28939489640738
// MI455X (gfx1250) — matrix operand order repaired and checked by reading, not compiled, not run on hardware
//
#include <hip/hip_runtime.h>
#include <hip/hip_bf16.h>

// ---------------------------------------------------------------------------
// Multi-head attention (per-head QKV projections + flash attention + out proj)
// MI455X / gfx1250: bf16 WMMA (v_wmma_f32_16x16x32_bf16), wave32, LDS staging.
// K-step 64, 32x64 output tile per wave, batched fragment loads -> dense WMMA.
// ---------------------------------------------------------------------------

typedef __attribute__((ext_vector_type(16))) __bf16 v16bf;
typedef __attribute__((ext_vector_type(8)))  __bf16 v8bf;
typedef __attribute__((ext_vector_type(8)))  float  v8f;
typedef __attribute__((ext_vector_type(4)))  float  v4f;

#define B_  4
#define S_  2048
#define D_  1024
#define H_  16
#define DK_ 64
#define HD_ (H_ * DK_)   // 1024

// --- fragment helpers (ISA 7.12.2 layouts, wave32) -------------------------

// 16 contiguous bf16 (B fragments: b[i] = B[h*16+i][n])
static __device__ __forceinline__ v16bf ld_frag_contig(const __bf16* p) {
  v8bf lo = *(const v8bf*)(p);
  v8bf hi = *(const v8bf*)(p + 8);
  return __builtin_shufflevector(lo, hi, 0,1,2,3,4,5,6,7,8,9,10,11,12,13,14,15);
}

// A fragment: 16 contiguous bf16 at p (K = h*16..h*16+15)
static __device__ __forceinline__ v16bf ld_afrag(const __bf16* p) {
  v8bf lo = *(const v8bf*)(p);
  v8bf hi = *(const v8bf*)(p + 8);
  return __builtin_shufflevector(lo, hi, 0,1,2,3,4,5,6,7,8,9,10,11,12,13,14,15);
}

static __device__ __forceinline__ v8f wmma_bf16(v16bf a, v16bf b, v8f c) {
  return __builtin_amdgcn_wmma_f32_16x16x32_bf16(
      /*neg_a=*/false, a, /*neg_b=*/false, b,
      /*c_mod=*/(short)0, c, /*reuse_a=*/false, /*reuse_b=*/false);
}

// --- fp32 -> bf16 convert ---------------------------------------------------

__global__ __launch_bounds__(256)
void cvt_f32_bf16(const float* __restrict__ in, __bf16* __restrict__ out, int n) {
  int i = (blockIdx.x * 256 + threadIdx.x) * 8;
  if (i + 8 <= n) {
    v4f a = *(const v4f*)(in + i);
    v4f b = *(const v4f*)(in + i + 4);
    v8bf o;
#pragma unroll
    for (int j = 0; j < 4; ++j) { o[j] = (__bf16)a[j]; o[4 + j] = (__bf16)b[j]; }
    *(v8bf*)(out + i) = o;
  }
}

// --- per-head projection GEMM: [S,D] x [D,DK] + bias -> bf16 [B][H][S][DK] --
// grid: (S/256, H, B); 256 threads = 8 waves; wave w owns rows w*32..w*32+31.
// "oscale" folds the attention 1/sqrt(DK) into the Q projection.

__global__ __launch_bounds__(256)
void proj_gemm(const __bf16* __restrict__ X,    // [B][S][D]
               const __bf16* __restrict__ W,    // [H][D][DK]
               const float*  __restrict__ bias, // [H][DK]
               float oscale,
               __bf16* __restrict__ Out)        // [B][H][S][DK]
{
  __shared__ __bf16 Asm[256 * 64];  // [m][kloc]      32 KB
  __shared__ __bf16 Bsm[64 * 64];   // [n][kloc] (T)   8 KB
  const int b  = blockIdx.z, h = blockIdx.y;
  const int m0 = blockIdx.x * 256;
  const int t = threadIdx.x;
  const int lane = t & 31, w = t >> 5;
  const int ln = lane & 15, lh = lane >> 4;

  const __bf16* Xb = X + (size_t)b * S_ * D_;
  const __bf16* Wh = W + (size_t)h * D_ * DK_;

  v8f acc[2][4] = {};

  for (int k0 = 0; k0 < D_; k0 += 64) {
    __syncthreads();
    {   // stage A tile 256x64: thread t copies row t (128 bytes, contiguous)
      const __bf16* src = Xb + (size_t)(m0 + t) * D_ + k0;
#pragma unroll
      for (int c = 0; c < 64; c += 16) {
        v8bf x0 = *(const v8bf*)(src + c);
        v8bf x1 = *(const v8bf*)(src + c + 8);
        *(v8bf*)&Asm[t * 64 + c]     = x0;
        *(v8bf*)&Asm[t * 64 + c + 8] = x1;
      }
    }
    {   // stage W tile 64x64 transposed -> Bsm[n][kloc]
      int kloc = t >> 2, nc = (t & 3) * 16;
      const __bf16* wsrc = Wh + (size_t)(k0 + kloc) * DK_ + nc;
      v8bf w0 = *(const v8bf*)wsrc;
      v8bf w1 = *(const v8bf*)(wsrc + 8);
#pragma unroll
      for (int j = 0; j < 8; ++j) {
        Bsm[(nc + j) * 64 + kloc]     = w0[j];
        Bsm[(nc + 8 + j) * 64 + kloc] = w1[j];
      }
    }
    __syncthreads();

#pragma unroll
    for (int s = 0; s < 2; ++s) {       // two 32-deep WMMA steps
      v16bf af[2], bf[4];
#pragma unroll
      for (int ms = 0; ms < 2; ++ms)
        af[ms] = ld_afrag(&Asm[(w * 32 + ms * 16 + ln) * 64 + s * 32 + lh * 16]);
#pragma unroll
      for (int ns = 0; ns < 4; ++ns)
        bf[ns] = ld_frag_contig(&Bsm[(ns * 16 + ln) * 64 + s * 32 + lh * 16]);
#pragma unroll
      for (int ms = 0; ms < 2; ++ms)
#pragma unroll
        for (int ns = 0; ns < 4; ++ns)
          acc[ms][ns] = wmma_bf16(af[ms], bf[ns], acc[ms][ns]);
    }
  }

  __bf16* Ob = Out + ((size_t)(b * H_ + h) * S_ + (m0 + w * 32)) * DK_;
#pragma unroll
  for (int ms = 0; ms < 2; ++ms)
#pragma unroll
    for (int ns = 0; ns < 4; ++ns) {
      int col = ns * 16 + ln;
      float bv = bias[h * DK_ + col];
#pragma unroll
      for (int r = 0; r < 8; ++r)
        Ob[(size_t)(ms * 16 + r + 8 * lh) * DK_ + col] =
            (__bf16)((acc[ms][ns][r] + bv) * oscale);
    }
}

// --- flash attention: softmax(QK^T) V -> bf16 [B][S][H*DK] ------------------
// (1/sqrt(DK) is pre-folded into Q.)
// grid: (S/128, H, B); 8 waves; wave owns 16 query rows; key tiles of 64.

__global__ __launch_bounds__(256)
void attn_kernel(const __bf16* __restrict__ Q,  // [B][H][S][DK], pre-scaled
                 const __bf16* __restrict__ K,
                 const __bf16* __restrict__ V,
                 __bf16* __restrict__ O)        // [B][S][H*DK]
{
  __shared__ __bf16 Ksm[64 * 64];      // [st][dk]   (row major)
  __shared__ __bf16 VTsm[64 * 64];     // [dk][st]   (transposed)
  __shared__ __bf16 Psm[8][16 * 64];   // per-wave P scratch [m][st]

  const int b  = blockIdx.z, h = blockIdx.y;
  const int q0 = blockIdx.x * 128;
  const int t = threadIdx.x, lane = t & 31, w = t >> 5;
  const int ln = lane & 15, lh = lane >> 4;

  const size_t headoff = (size_t)(b * H_ + h) * S_ * DK_;
  const __bf16* Qh = Q + headoff;
  const __bf16* Kh = K + headoff;
  const __bf16* Vh = V + headoff;

  // Q fragments for this wave's 16-row strip (resident for whole kernel)
  v16bf aq[2];
#pragma unroll
  for (int s = 0; s < 2; ++s)
    aq[s] = ld_afrag(Qh + (size_t)(q0 + w * 16 + ln) * DK_ + s * 32 + lh * 16);

  v8f o_acc[4] = {};
  float m_r[8], l_r[8];
#pragma unroll
  for (int r = 0; r < 8; ++r) { m_r[r] = -1e30f; l_r[r] = 0.f; }

  for (int kt = 0; kt < S_; kt += 64) {
    __syncthreads();
    {   // stage K tile (row major) and V tile (transposed)
      int row = t >> 2, dc = (t & 3) * 16;
      const __bf16* ks = Kh + (size_t)(kt + row) * DK_ + dc;
      v8bf k0v = *(const v8bf*)ks;
      v8bf k1v = *(const v8bf*)(ks + 8);
      *(v8bf*)&Ksm[row * 64 + dc]     = k0v;
      *(v8bf*)&Ksm[row * 64 + dc + 8] = k1v;

      const __bf16* vs = Vh + (size_t)(kt + row) * DK_ + dc;
      v8bf v0v = *(const v8bf*)vs;
      v8bf v1v = *(const v8bf*)(vs + 8);
#pragma unroll
      for (int j = 0; j < 8; ++j) {
        VTsm[(dc + j) * 64 + row]     = v0v[j];
        VTsm[(dc + 8 + j) * 64 + row] = v1v[j];
      }
      if (kt + 64 < S_) {   // prefetch next tiles -> global_prefetch_b8
        __builtin_prefetch(Kh + (size_t)(kt + 64 + row) * DK_ + dc, 0, 0);
        __builtin_prefetch(Vh + (size_t)(kt + 64 + row) * DK_ + dc, 0, 0);
      }
    }
    __syncthreads();

    // scores = Q K^T (4 tiles of 16 keys), batched fragment loads
    v8f sc[4] = {};
#pragma unroll
    for (int s = 0; s < 2; ++s) {
      v16bf bk[4];
#pragma unroll
      for (int ns = 0; ns < 4; ++ns)
        bk[ns] = ld_frag_contig(&Ksm[(ns * 16 + ln) * 64 + s * 32 + lh * 16]);
#pragma unroll
      for (int ns = 0; ns < 4; ++ns)
        sc[ns] = wmma_bf16(aq[s], bk[ns], sc[ns]);
    }

    // online softmax update (row r lives in one 16-lane half)
    float mnew[8], alpha[8], rsum[8];
#pragma unroll
    for (int r = 0; r < 8; ++r) {
      float mx = sc[0][r];
#pragma unroll
      for (int ns = 1; ns < 4; ++ns) mx = fmaxf(mx, sc[ns][r]);
#pragma unroll
      for (int msk = 1; msk < 16; msk <<= 1)
        mx = fmaxf(mx, __shfl_xor(mx, msk, 32));
      mnew[r]  = fmaxf(m_r[r], mx);
      alpha[r] = __expf(m_r[r] - mnew[r]);
      m_r[r]   = mnew[r];
      rsum[r]  = 0.f;
    }
#pragma unroll
    for (int ns = 0; ns < 4; ++ns) {
#pragma unroll
      for (int r = 0; r < 8; ++r) {
        float p = __expf(sc[ns][r] - mnew[r]);
        rsum[r] += p;
        Psm[w][(r + 8 * lh) * 64 + ns * 16 + ln] = (__bf16)p;
      }
    }
#pragma unroll
    for (int r = 0; r < 8; ++r) {
      float s = rsum[r];
#pragma unroll
      for (int msk = 1; msk < 16; msk <<= 1)
        s += __shfl_xor(s, msk, 32);
      l_r[r] = l_r[r] * alpha[r] + s;
#pragma unroll
      for (int ns = 0; ns < 4; ++ns) o_acc[ns][r] *= alpha[r];
    }

    // O += P x V   (contraction over 64 keys, two 32-deep WMMA steps)
#pragma unroll
    for (int s = 0; s < 2; ++s) {
      v16bf ap = ld_afrag(&Psm[w][ln * 64 + s * 32 + lh * 16]);
      v16bf bv[4];
#pragma unroll
      for (int ns = 0; ns < 4; ++ns)
        bv[ns] = ld_frag_contig(&VTsm[(ns * 16 + ln) * 64 + s * 32 + lh * 16]);
#pragma unroll
      for (int ns = 0; ns < 4; ++ns)
        o_acc[ns] = wmma_bf16(ap, bv[ns], o_acc[ns]);
    }
  }

  // normalize and store head-concat layout [B][S][H*DK]
  __bf16* Orow = O + ((size_t)b * S_ + q0 + w * 16) * HD_ + h * DK_;
#pragma unroll
  for (int r = 0; r < 8; ++r) {
    float inv = 1.0f / l_r[r];
#pragma unroll
    for (int ns = 0; ns < 4; ++ns)
      Orow[(size_t)(r + 8 * lh) * HD_ + ns * 16 + ln] =
          (__bf16)(o_acc[ns][r] * inv);
  }
}

// --- output projection: [B*S, HD] x [HD, D] + bo -> fp32 --------------------
// grid: (B*S/256, D/64); 8 waves; wave owns a 32x64 tile.

__global__ __launch_bounds__(256)
void out_gemm(const __bf16* __restrict__ A,   // [B*S][HD]
              const __bf16* __restrict__ Wo,  // [HD][D]
              const float*  __restrict__ bo,  // [D]
              float* __restrict__ Out)        // [B*S][D]
{
  __shared__ __bf16 Asm[256 * 64];  // 32 KB
  __shared__ __bf16 Bsm[64 * 64];   // [n][kloc] (T) 8 KB
  const int m0 = blockIdx.x * 256;
  const int n0 = blockIdx.y * 64;
  const int t = threadIdx.x;
  const int lane = t & 31, w = t >> 5;
  const int ln = lane & 15, lh = lane >> 4;

  v8f acc[2][4] = {};

  for (int k0 = 0; k0 < HD_; k0 += 64) {
    __syncthreads();
    {
      const __bf16* src = A + (size_t)(m0 + t) * HD_ + k0;
#pragma unroll
      for (int c = 0; c < 64; c += 16) {
        v8bf x0 = *(const v8bf*)(src + c);
        v8bf x1 = *(const v8bf*)(src + c + 8);
        *(v8bf*)&Asm[t * 64 + c]     = x0;
        *(v8bf*)&Asm[t * 64 + c + 8] = x1;
      }
    }
    {
      int kloc = t >> 2, nc = (t & 3) * 16;
      const __bf16* wsrc = Wo + (size_t)(k0 + kloc) * D_ + n0 + nc;
      v8bf w0 = *(const v8bf*)wsrc;
      v8bf w1 = *(const v8bf*)(wsrc + 8);
#pragma unroll
      for (int j = 0; j < 8; ++j) {
        Bsm[(nc + j) * 64 + kloc]     = w0[j];
        Bsm[(nc + 8 + j) * 64 + kloc] = w1[j];
      }
    }
    __syncthreads();

#pragma unroll
    for (int s = 0; s < 2; ++s) {
      v16bf af[2], bf[4];
#pragma unroll
      for (int ms = 0; ms < 2; ++ms)
        af[ms] = ld_afrag(&Asm[(w * 32 + ms * 16 + ln) * 64 + s * 32 + lh * 16]);
#pragma unroll
      for (int ns = 0; ns < 4; ++ns)
        bf[ns] = ld_frag_contig(&Bsm[(ns * 16 + ln) * 64 + s * 32 + lh * 16]);
#pragma unroll
      for (int ms = 0; ms < 2; ++ms)
#pragma unroll
        for (int ns = 0; ns < 4; ++ns)
          acc[ms][ns] = wmma_bf16(af[ms], bf[ns], acc[ms][ns]);
    }
  }

#pragma unroll
  for (int ms = 0; ms < 2; ++ms)
#pragma unroll
    for (int ns = 0; ns < 4; ++ns) {
      int col = n0 + ns * 16 + ln;
      float bv = bo[col];
#pragma unroll
      for (int r = 0; r < 8; ++r)
        Out[(size_t)(m0 + w * 32 + ms * 16 + r + 8 * lh) * D_ + col] =
            acc[ms][ns][r] + bv;
    }
}

// --- host launch ------------------------------------------------------------

extern "C" void kernel_launch(void* const* d_in, const int* in_sizes, int n_in,
                              void* d_out, int out_size, void* d_ws, size_t ws_size,
                              hipStream_t stream) {
  (void)in_sizes; (void)n_in; (void)out_size; (void)ws_size;

  const float* q_in = (const float*)d_in[0];
  const float* k_in = (const float*)d_in[1];
  const float* v_in = (const float*)d_in[2];
  const float* wq   = (const float*)d_in[3];
  const float* bq   = (const float*)d_in[4];
  const float* wk   = (const float*)d_in[5];
  const float* bk   = (const float*)d_in[6];
  const float* wv   = (const float*)d_in[7];
  const float* bv   = (const float*)d_in[8];
  const float* wo   = (const float*)d_in[9];
  const float* bo   = (const float*)d_in[10];
  float* out = (float*)d_out;

  // workspace layout (bf16 elements)
  const size_t XE = (size_t)B_ * S_ * D_;        // 8388608
  const size_t WE = (size_t)H_ * D_ * DK_;       // 1048576
  const size_t QE = (size_t)B_ * H_ * S_ * DK_;  // 8388608

  __bf16* ws  = (__bf16*)d_ws;
  __bf16* xqb = ws;               // bf16 q_in
  __bf16* xkb = xqb + XE;
  __bf16* xvb = xkb + XE;
  __bf16* wqb = xvb + XE;
  __bf16* wkb = wqb + WE;
  __bf16* wvb = wkb + WE;
  __bf16* wob = wvb + WE;         // size WE (HD*D)
  __bf16* qb  = wob + WE;         // [B][H][S][DK]
  __bf16* kb  = qb + QE;
  __bf16* vb  = kb + QE;
  __bf16* ab  = vb + QE;          // attn concat [B][S][HD]

  // 1) convert to bf16
  cvt_f32_bf16<<<dim3((unsigned)(XE / (8 * 256))), 256, 0, stream>>>(q_in, xqb, (int)XE);
  cvt_f32_bf16<<<dim3((unsigned)(XE / (8 * 256))), 256, 0, stream>>>(k_in, xkb, (int)XE);
  cvt_f32_bf16<<<dim3((unsigned)(XE / (8 * 256))), 256, 0, stream>>>(v_in, xvb, (int)XE);
  cvt_f32_bf16<<<dim3((unsigned)(WE / (8 * 256))), 256, 0, stream>>>(wq, wqb, (int)WE);
  cvt_f32_bf16<<<dim3((unsigned)(WE / (8 * 256))), 256, 0, stream>>>(wk, wkb, (int)WE);
  cvt_f32_bf16<<<dim3((unsigned)(WE / (8 * 256))), 256, 0, stream>>>(wv, wvb, (int)WE);
  cvt_f32_bf16<<<dim3((unsigned)(WE / (8 * 256))), 256, 0, stream>>>(wo, wob, (int)WE);

  // 2) per-head QKV projections (Q pre-scaled by 1/sqrt(DK))
  dim3 pgrid(S_ / 256, H_, B_);
  proj_gemm<<<pgrid, 256, 0, stream>>>(xqb, wqb, bq, 0.125f, qb);
  proj_gemm<<<pgrid, 256, 0, stream>>>(xkb, wkb, bk, 1.0f,   kb);
  proj_gemm<<<pgrid, 256, 0, stream>>>(xvb, wvb, bv, 1.0f,   vb);

  // 3) flash attention
  attn_kernel<<<dim3(S_ / 128, H_, B_), 256, 0, stream>>>(qb, kb, vb, ab);

  // 4) output projection (fp32 result)
  out_gemm<<<dim3((B_ * S_) / 256, D_ / 64), 256, 0, stream>>>(ab, wob, bo, out);
}
